// semantic_80393197847173
// MI455X (gfx1250) — compile-verified, matrix operand order checked by reading, not run on hardware
//
#include <hip/hip_runtime.h>
#include <hip/hip_bf16.h>
#include <math.h>

// Problem constants (match reference)
constexpr int Bn = 2, T = 128, Cc = 527, DA = 1024, DW = 300, H = 1024;
constexpr int BT  = Bn * T;   // 256
constexpr int Cp  = 528;      // C padded to multiple of 16
constexpr int Kp2 = 320;      // DW padded to multiple of 32

typedef __attribute__((ext_vector_type(16))) _Float16 v16h;
typedef __attribute__((ext_vector_type(8)))  _Float16 v8h;
typedef __attribute__((ext_vector_type(8)))  float    v8f;
typedef __attribute__((ext_vector_type(4)))  float    v4f;

// ---------------------------------------------------------------------------
// Fragment load for V_WMMA_F32_16X16X32_F16 (wave32), f16 source, row-K.
// Lane l<16 -> M/N = l, elems[0..7] = K(k0..k0+7), elems[8..15] = K(k0+16..23);
// lane l+16 -> K-octets shifted by +8.  p must point at row base + k0 + halfoff.
// Two aligned 16B loads -> global_load_b128.
// ---------------------------------------------------------------------------
__device__ inline v16h frag_ld(const _Float16* __restrict__ p) {
  v8h lo = *(const v8h*)(p);
  v8h hi = *(const v8h*)(p + 16);
  return __builtin_shufflevector(lo, hi, 0, 1, 2, 3, 4, 5, 6, 7,
                                 8, 9, 10, 11, 12, 13, 14, 15);
}

// ---------------------------------------------------------------------------
// Staging: f32 -> f16 copy with zero row/col padding (one thread per dst elem)
// ---------------------------------------------------------------------------
__global__ void k_cvt_pad(const float* __restrict__ src, int srows, int scols,
                          _Float16* __restrict__ dst, long total, int dcols) {
  long id = (long)blockIdx.x * blockDim.x + threadIdx.x;
  if (id >= total) return;
  int r = (int)(id / dcols), c = (int)(id % dcols);
  float x = (r < srows && c < scols) ? src[(size_t)r * scols + c] : 0.0f;
  dst[id] = (_Float16)x;
}

// Staging: audio [Bn,T,DA] f32 -> audioT [Bn,DA,T] f16
__global__ void k_transpose_cvt(const float* __restrict__ audio,
                                _Float16* __restrict__ dstT) {
  long id = (long)blockIdx.x * blockDim.x + threadIdx.x;  // coalesced read
  if (id >= (long)Bn * T * DA) return;
  int d = (int)(id % DA);
  int t = (int)((id / DA) % T);
  int b = (int)(id / ((long)DA * T));
  dstT[((size_t)b * DA + d) * T + t] = (_Float16)audio[id];
}

// ---------------------------------------------------------------------------
// v[h] = sum_o Wa[o] * W3[o,h]  (fc_3/fc_a fused; biases cancel under softmax)
// ---------------------------------------------------------------------------
__global__ void k_fuse_v(const float* __restrict__ W3, const float* __restrict__ Wa,
                         float* __restrict__ v) {
  int h = blockIdx.x * blockDim.x + threadIdx.x;
  float s = 0.0f;
  for (int o = 0; o < H; ++o) s = fmaf(Wa[o], W3[(size_t)o * H + h], s);
  v[h] = s;
}

// ---------------------------------------------------------------------------
// C[M,N] = A[M,K] * B[N,K]^T, f16 operands, f32 result. One wave / 16x16 tile.
// All rows padded; K multiple of 32; no guards in the hot loop.
// ---------------------------------------------------------------------------
__global__ void k_gemm_nt(const _Float16* __restrict__ A, int lda,
                          const _Float16* __restrict__ B, int ldb,
                          float* __restrict__ Cm, int ldc,
                          int Ntiles, int K) {
  const int lane = threadIdx.x & 31;
  const int wv = blockIdx.x * (blockDim.x >> 5) + (threadIdx.x >> 5);
  const int tm = wv / Ntiles, tn = wv % Ntiles;
  const int halfoff = ((lane >> 4) & 1) << 3;
  const _Float16* pa = A + (size_t)(tm * 16 + (lane & 15)) * lda + halfoff;
  const _Float16* pb = B + (size_t)(tn * 16 + (lane & 15)) * ldb + halfoff;

  v8f acc = {};
  for (int k0 = 0; k0 < K; k0 += 32) {
    v16h af = frag_ld(pa + k0);
    v16h bf = frag_ld(pb + k0);
    acc = __builtin_amdgcn_wmma_f32_16x16x32_f16(false, af, false, bf,
                                                 (short)0, acc, false, false);
  }
  const int col = tn * 16 + (lane & 15);
#pragma unroll
  for (int r = 0; r < 8; ++r) {
    int row = tm * 16 + r + (((lane >> 4) & 1) << 3);
    Cm[(size_t)row * ldc + col] = acc[r];
  }
}

// ---------------------------------------------------------------------------
// score[b,c,t] = sum_h tanh(a[bt,h] * w[c,h]) * v[h]
// One wave per (bt,c); float4 loads, L2-resident operands.
// ---------------------------------------------------------------------------
__global__ void k_score(const float* __restrict__ a, const float* __restrict__ w,
                        const float* __restrict__ v, float* __restrict__ score) {
  const int lane = threadIdx.x & 31;
  const int wv = blockIdx.x * (blockDim.x >> 5) + (threadIdx.x >> 5);
  const int bt = wv / Cp;
  const int c  = wv % Cp;
  const float* ar = a + (size_t)bt * H;
  const float* wr = w + (size_t)c * H;
  float s = 0.0f;
#pragma unroll 2
  for (int it = 0; it < H / 128; ++it) {
    int j = it * 128 + lane * 4;
    v4f av = *(const v4f*)(ar + j);
    v4f wf = *(const v4f*)(wr + j);
    v4f vf = *(const v4f*)(v + j);
#pragma unroll
    for (int e = 0; e < 4; ++e)
      s = fmaf(tanhf(av[e] * wf[e]), vf[e], s);
  }
#pragma unroll
  for (int off = 16; off; off >>= 1) s += __shfl_xor(s, off, 32);
  if (lane == 0) {
    int b = bt / T, t = bt % T;
    score[((size_t)b * Cp + c) * T + t] = s;
  }
}

// ---------------------------------------------------------------------------
// softmax over T per (b,c); writes coef directly as f16 for the pooling GEMM.
// ---------------------------------------------------------------------------
__global__ void k_softmax(const float* __restrict__ score,
                          _Float16* __restrict__ coef_h) {
  const int lane = threadIdx.x & 31;
  const int wv = blockIdx.x * (blockDim.x >> 5) + (threadIdx.x >> 5);  // b*Cp + c
  const float* s = score + (size_t)wv * T;
  float x[4], m = -3.0e38f;
#pragma unroll
  for (int j = 0; j < 4; ++j) { x[j] = s[lane + 32 * j]; m = fmaxf(m, x[j]); }
#pragma unroll
  for (int off = 16; off; off >>= 1) m = fmaxf(m, __shfl_xor(m, off, 32));
  float sum = 0.0f;
#pragma unroll
  for (int j = 0; j < 4; ++j) { x[j] = __expf(x[j] - m); sum += x[j]; }
#pragma unroll
  for (int off = 16; off; off >>= 1) sum += __shfl_xor(sum, off, 32);
  const float inv = 1.0f / sum;
  _Float16* cf = coef_h + (size_t)wv * T;
#pragma unroll
  for (int j = 0; j < 4; ++j) cf[lane + 32 * j] = (_Float16)(x[j] * inv);
}

// ---------------------------------------------------------------------------
// out[b,c,d] = sum_t coef[b,c,t] * audio[b,t,d]
// A = coef_h[b] [Cp,T] row-K; B = audioT_h[b] [DA,T] row-K. K = T = 128.
// ---------------------------------------------------------------------------
__global__ void k_pool(const _Float16* __restrict__ coef_h,
                       const _Float16* __restrict__ audioT_h,
                       float* __restrict__ out) {
  const int lane = threadIdx.x & 31;
  const int wv = blockIdx.x * (blockDim.x >> 5) + (threadIdx.x >> 5);
  constexpr int tilesM = Cp / 16;   // 33
  constexpr int tilesN = DA / 16;   // 64
  const int b = wv / (tilesM * tilesN);
  const int r0 = wv % (tilesM * tilesN);
  const int tm = r0 / tilesN, tn = r0 % tilesN;
  const int halfoff = ((lane >> 4) & 1) << 3;
  const _Float16* pa = coef_h + (size_t)b * Cp * T
                       + (size_t)(tm * 16 + (lane & 15)) * T + halfoff;
  const _Float16* pb = audioT_h + (size_t)b * DA * T
                       + (size_t)(tn * 16 + (lane & 15)) * T + halfoff;

  v8f acc = {};
#pragma unroll
  for (int k0 = 0; k0 < T; k0 += 32) {
    v16h af = frag_ld(pa + k0);
    v16h bf = frag_ld(pb + k0);
    acc = __builtin_amdgcn_wmma_f32_16x16x32_f16(false, af, false, bf,
                                                 (short)0, acc, false, false);
  }
  const int col = tn * 16 + (lane & 15);
#pragma unroll
  for (int r = 0; r < 8; ++r) {
    int row = tm * 16 + r + (((lane >> 4) & 1) << 3);
    if (row < Cc) out[((size_t)b * Cc + row) * DA + col] = acc[r];
  }
}

// ---------------------------------------------------------------------------
extern "C" void kernel_launch(void* const* d_in, const int* in_sizes, int n_in,
                              void* d_out, int out_size, void* d_ws, size_t ws_size,
                              hipStream_t stream) {
  (void)in_sizes; (void)n_in; (void)out_size; (void)ws_size;
  const float* audio = (const float*)d_in[0];  // [B,T,DA]
  const float* word  = (const float*)d_in[1];  // [C,DW]
  const float* W1    = (const float*)d_in[2];  // [H,DA]
  const float* W2    = (const float*)d_in[3];  // [H,DW]
  const float* W3    = (const float*)d_in[4];  // [H,H]
  const float* Wa    = (const float*)d_in[6];  // [1,H]
  // b3 (d_in[5]) and ba (d_in[7]) cancel under softmax shift invariance.
  float* out = (float*)d_out;                  // [B,C,DA]

  // ---- workspace carve (all segment sizes multiples of 16 bytes) ----------
  char* ws = (char*)d_ws;
  float* v      = (float*)ws;              ws += (size_t)H * 4;            // 4 KB
  float* aBT    = (float*)ws;              ws += (size_t)BT * H * 4;       // 1 MB
  float* wC     = (float*)ws;              ws += (size_t)Cp * H * 4;       // 2.1 MB
  float* score  = (float*)ws;              ws += (size_t)Bn * Cp * T * 4;  // 540 KB
  _Float16* audio_h  = (_Float16*)ws;      ws += (size_t)BT * DA * 2;      // 512 KB
  _Float16* W1_h     = (_Float16*)ws;      ws += (size_t)H * DA * 2;       // 2 MB
  _Float16* word_h   = (_Float16*)ws;      ws += (size_t)Cp * Kp2 * 2;     // 330 KB
  _Float16* W2_h     = (_Float16*)ws;      ws += (size_t)H * Kp2 * 2;      // 640 KB
  _Float16* audioT_h = (_Float16*)ws;      ws += (size_t)Bn * DA * T * 2;  // 512 KB
  _Float16* coef_h   = (_Float16*)ws;      ws += (size_t)Bn * Cp * T * 2;  // 264 KB

  const int thr = 256;  // 8 waves (wave32)
  auto blocks = [](long n, int t) { return (int)((n + t - 1) / t); };

  // ---- staging (f32 -> f16, padding, transpose) ---------------------------
  long n;
  n = (long)BT * DA;
  k_cvt_pad<<<blocks(n, thr), thr, 0, stream>>>(audio, BT, DA, audio_h, n, DA);
  n = (long)H * DA;
  k_cvt_pad<<<blocks(n, thr), thr, 0, stream>>>(W1, H, DA, W1_h, n, DA);
  n = (long)Cp * Kp2;
  k_cvt_pad<<<blocks(n, thr), thr, 0, stream>>>(word, Cc, DW, word_h, n, Kp2);
  n = (long)H * Kp2;
  k_cvt_pad<<<blocks(n, thr), thr, 0, stream>>>(W2, H, DW, W2_h, n, Kp2);
  n = (long)Bn * T * DA;
  k_transpose_cvt<<<blocks(n, thr), thr, 0, stream>>>(audio, audioT_h);

  // ---- fused readout vector ----------------------------------------------
  k_fuse_v<<<H / thr, thr, 0, stream>>>(W3, Wa, v);

  // ---- GEMM1: a = audio @ W1^T   (16x64 tiles, K=1024) --------------------
  k_gemm_nt<<<(16 * 64) / 8, thr, 0, stream>>>(audio_h, DA, W1_h, DA,
                                               aBT, H, 64, DA);
  // ---- GEMM2: w = word @ W2^T    (33x64 tiles, K=320 padded) --------------
  k_gemm_nt<<<(33 * 64) / 8, thr, 0, stream>>>(word_h, Kp2, W2_h, Kp2,
                                               wC, H, 64, Kp2);

  // ---- scores (tanh path, fp32) ------------------------------------------
  k_score<<<(BT * Cp) / 8, thr, 0, stream>>>(aBT, wC, v, score);

  // ---- softmax over T -> f16 coef ----------------------------------------
  k_softmax<<<(Bn * Cp) / 8, thr, 0, stream>>>(score, coef_h);

  // ---- softmax-weighted pooling (WMMA over K=T) ---------------------------
  k_pool<<<(Bn * 33 * 64) / 8, thr, 0, stream>>>(coef_h, audioT_h, out);
}
